// RWKV7Attention_13752485282225
// MI455X (gfx1250) — matrix operand order repaired and checked by reading, not compiled, not run on hardware
//
#include <hip/hip_runtime.h>

typedef __attribute__((ext_vector_type(16))) _Float16 v16h;
typedef __attribute__((ext_vector_type(8)))  _Float16 h8;
typedef __attribute__((ext_vector_type(8)))  float    v8f;

namespace {
constexpr int B_  = 8;
constexpr int T_  = 2048;
constexpr int H_  = 1024;
constexpr int NH_ = 16;
constexpr int DK_ = 64;
constexpr int DV_ = 64;
constexpr int VD_ = 1024;
constexpr int BT_ = B_ * T_;
constexpr float W_SCALE_ = -0.6065306597126334f;
constexpr float GN_EPS_  = 64.0f * 1e-5f;
}

__device__ __forceinline__ float sigmoidf_(float x) {
  return 1.0f / (1.0f + __expf(-x));
}

__device__ __forceinline__ float warp_sum(float v) {
#pragma unroll
  for (int m = 16; m >= 1; m >>= 1) v += __shfl_xor(v, m, 32);
  return v;
}

// ---------------------------------------------------------------- elementwise
__global__ void cvt_f16_kernel(const float* __restrict__ src,
                               _Float16* __restrict__ dst, int n) {
  int i = blockIdx.x * blockDim.x + threadIdx.x;
  if (i < n) dst[i] = (_Float16)src[i];
}

__global__ void act_cvt_f16_kernel(const float* __restrict__ src,
                                   _Float16* __restrict__ dst, int n, int do_tanh) {
  int i = blockIdx.x * blockDim.x + threadIdx.x;
  if (i < n) {
    float v = src[i];
    if (do_tanh) v = tanhf(v);
    dst[i] = (_Float16)v;
  }
}

// token shift + six mixes, output f16
__global__ void prep_mix_kernel(const float* __restrict__ x,
                                const float* __restrict__ mr, const float* __restrict__ mw,
                                const float* __restrict__ mk, const float* __restrict__ mv,
                                const float* __restrict__ ma, const float* __restrict__ mg,
                                _Float16* __restrict__ xr, _Float16* __restrict__ xw,
                                _Float16* __restrict__ xk, _Float16* __restrict__ xv,
                                _Float16* __restrict__ xa, _Float16* __restrict__ xg) {
  int i = blockIdx.x * blockDim.x + threadIdx.x;   // over BT_*H_
  int bt = i / H_;
  int ch = i & (H_ - 1);
  float cur  = x[i];
  float prev = ((bt % T_) == 0) ? 0.0f : x[i - H_];
  float d = prev - cur;
  xr[i] = (_Float16)(cur + d * mr[ch]);
  xw[i] = (_Float16)(cur + d * mw[ch]);
  xk[i] = (_Float16)(cur + d * mk[ch]);
  xv[i] = (_Float16)(cur + d * mv[ch]);
  xa[i] = (_Float16)(cur + d * ma[ch]);
  xg[i] = (_Float16)(cur + d * mg[ch]);
}

// ---------------------------------------------------------------- WMMA GEMM
// C[M,N] = A[M,K] * W[N,K]^T   (A,W f16 row-major, C f32 row-major)
// block: 256 thr = 8 waves; wave -> 16 x (NT*16) strip.
// N, K are compile-time: tile strides / k-offsets fold into load-store
// immediates (minimal per-iteration address VALU, lower reg pressure).
// Ping-pong double buffering unrolled by two 32-wide k-steps with in-place
// fragment reloads; final reloads overrun K by <= 64 halfs (operands live
// mid-workspace, harmless). Requires: M % 128 == 0, N % (NT*16) == 0,
// K % 32 == 0, K >= 64.
template <int NT, int N, int K>
__global__ void __launch_bounds__(256) gemm_f16_wmma(
    const _Float16* __restrict__ A, const _Float16* __restrict__ W,
    float* __restrict__ C, int M) {
  constexpr bool ODD = ((K / 32) & 1) != 0;
  const int lane = threadIdx.x & 31;
  const int wave = threadIdx.x >> 5;
  const int m0 = blockIdx.y * 128 + wave * 16;
  const int nbase = blockIdx.x * (NT * 16);
  const int rc = lane & 15;     // A row within tile / B,C col within tile
  const int hf = lane >> 4;     // K-half select / C row-half select
  (void)M;

  v8f acc[NT];
#pragma unroll
  for (int t = 0; t < NT; ++t)
#pragma unroll
    for (int e = 0; e < 8; ++e) acc[t][e] = 0.0f;

  const _Float16* Ab = A + (size_t)(m0 + rc) * K;
  const _Float16* Wb = W + (size_t)(nbase + rc) * K + hf * 16;

  auto loadA = [&](int k0) -> v16h {
    return *(const v16h*)(Ab + k0 + hf * 16);
  };

  // prologue: steps 0 and 1
  v16h af0 = loadA(0);
  v16h bf0[NT];
#pragma unroll
  for (int t = 0; t < NT; ++t) bf0[t] = *(const v16h*)(Wb + t * 16 * K);
  v16h af1 = loadA(32);
  v16h bf1[NT];
#pragma unroll
  for (int t = 0; t < NT; ++t) bf1[t] = *(const v16h*)(Wb + t * 16 * K + 32);

#pragma unroll 1
  for (int k0 = 0; k0 + 64 <= K; k0 += 64) {
    // phase 0: consume buf0 (k0), refill buf0 from k0+64 in place
#pragma unroll
    for (int t = 0; t < NT; ++t)
      acc[t] = __builtin_amdgcn_wmma_f32_16x16x32_f16(
          false, af0, false, bf0[t], (short)0, acc[t], false, false);
    af0 = loadA(k0 + 64);
#pragma unroll
    for (int t = 0; t < NT; ++t)
      bf0[t] = *(const v16h*)(Wb + t * 16 * K + k0 + 64);
    __builtin_prefetch((const void*)(Ab + k0 + 576), 0, 3);
    // phase 1: consume buf1 (k0+32), refill buf1 from k0+96 in place
#pragma unroll
    for (int t = 0; t < NT; ++t)
      acc[t] = __builtin_amdgcn_wmma_f32_16x16x32_f16(
          false, af1, false, bf1[t], (short)0, acc[t], false, false);
    af1 = loadA(k0 + 96);
#pragma unroll
    for (int t = 0; t < NT; ++t)
      bf1[t] = *(const v16h*)(Wb + t * 16 * K + k0 + 96);
  }
  if constexpr (ODD) {  // odd number of 32-steps (K=160): last step in buf0
#pragma unroll
    for (int t = 0; t < NT; ++t)
      acc[t] = __builtin_amdgcn_wmma_f32_16x16x32_f16(
          false, af0, false, bf0[t], (short)0, acc[t], false, false);
  }

  float* Cb = C + (size_t)m0 * N + nbase + rc;
#pragma unroll
  for (int t = 0; t < NT; ++t) {
#pragma unroll
    for (int e = 0; e < 8; ++e)
      Cb[(size_t)(e + hf * 8) * N + t * 16] = acc[t][e];
  }
}

template <int NT, int N, int K>
static void launch_gemm(const _Float16* A, const _Float16* W, float* C, int M,
                        hipStream_t stream) {
  dim3 grid((unsigned)(N / (NT * 16)), (unsigned)(M / 128));
  gemm_f16_wmma<NT, N, K><<<grid, dim3(256), 0, stream>>>(A, W, C, M);
}

// ---------------------------------------------------------------- gates
// wl -> w (log-decay), al -> a, gl -> g, v(raw in place) -> blended v
__global__ void epilogue_gates_kernel(float* __restrict__ wl, float* __restrict__ al,
                                      const float* __restrict__ vl, float* __restrict__ gl,
                                      float* __restrict__ v,
                                      const float* __restrict__ vfirst,
                                      const float* __restrict__ wbias,
                                      const float* __restrict__ abias,
                                      const float* __restrict__ vbias) {
  int i = blockIdx.x * blockDim.x + threadIdx.x;   // over BT_*H_
  int ch = i & (H_ - 1);
  wl[i] = W_SCALE_ * sigmoidf_(wl[i] + wbias[ch]);
  al[i] = sigmoidf_(al[i] + abias[ch]);
  float vs = sigmoidf_(vl[i] + vbias[ch]);
  float vr = v[i];
  v[i] = vr + (vfirst[i] - vr) * vs;
  gl[i] = sigmoidf_(gl[i]);
}

// per-head: kk = l2norm(k*k_k); adir=-kk; bdir=kk*a; k -> k*(1+(a-1)*k_a)
__global__ void kk_dirs_kernel(float* __restrict__ k, const float* __restrict__ a,
                               float* __restrict__ adir, float* __restrict__ bdir,
                               const float* __restrict__ k_k,
                               const float* __restrict__ k_a) {
  int warp = threadIdx.x >> 5;
  int lane = threadIdx.x & 31;
  int idx = blockIdx.x * 8 + warp;     // (bt,head)
  int h = idx % NH_;
  size_t base = (size_t)(idx / NH_) * H_ + (size_t)h * DK_;
  int c0 = lane, c1 = lane + 32;
  int hc0 = h * DK_ + c0, hc1 = h * DK_ + c1;
  float k0v = k[base + c0], k1v = k[base + c1];
  float kk0 = k0v * k_k[hc0], kk1 = k1v * k_k[hc1];
  float ss = warp_sum(kk0 * kk0 + kk1 * kk1);
  float inv = 1.0f / fmaxf(sqrtf(ss), 1e-12f);
  kk0 *= inv; kk1 *= inv;
  float a0 = a[base + c0], a1 = a[base + c1];
  adir[base + c0] = -kk0;          adir[base + c1] = -kk1;
  bdir[base + c0] = kk0 * a0;      bdir[base + c1] = kk1 * a1;
  k[base + c0] = k0v * (1.0f + (a0 - 1.0f) * k_a[hc0]);
  k[base + c1] = k1v * (1.0f + (a1 - 1.0f) * k_a[hc1]);
}

// ---------------------------------------------------------------- recurrence
// one block (256 thr) per (batch,head). 4 threads per value-column, each
// owning 16 S-rows in VGPRs; k-reduction closed with two shfl_xor hops
// (q = lane bits [4:3]). Next step's vectors are prefetched into registers
// while the current step computes (loader role tid>>6 is wave-uniform).
__global__ void __launch_bounds__(256) rwkv7_recurrence_kernel(
    const float* __restrict__ r, const float* __restrict__ w,
    const float* __restrict__ k, const float* __restrict__ v,
    const float* __restrict__ adir, const float* __restrict__ bdir,
    float* __restrict__ o) {
  const int bh = blockIdx.x;
  const int b = bh / NH_;
  const int h = bh % NH_;
  const int tid = threadIdx.x;
  const int lane = tid & 31;
  const int wave = tid >> 5;
  const int q = (lane >> 3) & 3;            // k-quarter, lane bits [4:3]
  const int col = wave * 8 + (lane & 7);    // value column 0..63
  const int vec = tid >> 6;                 // loader role 0..3 (wave-uniform)
  const int vidx = tid & 63;
  const int kq = q * 16;

  __shared__ float sr[DK_], se[DK_], sk[DK_], sa[DK_], sb[DK_];

  float S[16];
#pragma unroll
  for (int i = 0; i < 16; ++i) S[i] = 0.0f;

  const float* vp = (vec == 0) ? r : (vec == 1) ? w : (vec == 2) ? k : adir;

  size_t base = ((size_t)(b * T_)) * H_ + (size_t)h * DK_;
  float c_main = vp[base + vidx];
  float c_b    = (vec == 0) ? bdir[base + vidx] : 0.0f;
  float c_v    = v[base + col];

  for (int t = 0; t < T_; ++t) {
    // prefetch next step (last iteration re-reads current step harmlessly)
    size_t nbase = base + ((t + 1 < T_) ? (size_t)H_ : 0);
    float n_main = vp[nbase + vidx];
    float n_b    = (vec == 0) ? bdir[nbase + vidx] : 0.0f;
    float n_v    = v[nbase + col];

    if (vec == 0)      { sr[vidx] = c_main; sb[vidx] = c_b; }
    else if (vec == 1) { se[vidx] = __expf(c_main); }
    else if (vec == 2) { sk[vidx] = c_main; }
    else               { sa[vidx] = c_main; }
    __syncthreads();

    float sav = 0.0f;
#pragma unroll
    for (int i = 0; i < 16; ++i) sav += sa[kq + i] * S[i];
    sav += __shfl_xor(sav, 8, 32);
    sav += __shfl_xor(sav, 16, 32);

    float ov = 0.0f;
#pragma unroll
    for (int i = 0; i < 16; ++i) {
      float s = S[i] * se[kq + i] + sb[kq + i] * sav + sk[kq + i] * c_v;
      S[i] = s;
      ov += sr[kq + i] * s;
    }
    ov += __shfl_xor(ov, 8, 32);
    ov += __shfl_xor(ov, 16, 32);
    if (q == 0) o[base + col] = ov;
    __syncthreads();

    c_main = n_main; c_b = n_b; c_v = n_v;
    base = nbase;
  }
}

// ---------------------------------------------------------------- norm+bonus
__global__ void norm_bonus_gate_kernel(const float* __restrict__ o,
                                       const float* __restrict__ r,
                                       const float* __restrict__ k,
                                       const float* __restrict__ v,
                                       const float* __restrict__ g,
                                       const float* __restrict__ r_k,
                                       const float* __restrict__ gnw,
                                       const float* __restrict__ gnb,
                                       _Float16* __restrict__ out16) {
  int warp = threadIdx.x >> 5;
  int lane = threadIdx.x & 31;
  int idx = blockIdx.x * 8 + warp;     // (bt,head)
  int h = idx % NH_;
  size_t base = (size_t)(idx / NH_) * H_ + (size_t)h * DV_;
  int c0 = lane, c1 = lane + 32;
  int hc0 = h * DK_ + c0, hc1 = h * DK_ + c1;
  float o0 = o[base + c0], o1 = o[base + c1];
  float mean = warp_sum(o0 + o1) * (1.0f / DV_);
  float var  = warp_sum(o0 * o0 + o1 * o1) * (1.0f / DV_) - mean * mean;
  float rstd = rsqrtf(var + GN_EPS_);
  float s = warp_sum(r[base + c0] * k[base + c0] * r_k[hc0] +
                     r[base + c1] * k[base + c1] * r_k[hc1]);
  float on0 = (o0 - mean) * rstd * gnw[hc0] + gnb[hc0];
  float on1 = (o1 - mean) * rstd * gnw[hc1] + gnb[hc1];
  out16[base + c0] = (_Float16)((on0 + s * v[base + c0]) * g[base + c0]);
  out16[base + c1] = (_Float16)((on1 + s * v[base + c1]) * g[base + c1]);
}

// ---------------------------------------------------------------- launcher
extern "C" void kernel_launch(void* const* d_in, const int* in_sizes, int n_in,
                              void* d_out, int out_size, void* d_ws, size_t ws_size,
                              hipStream_t stream) {
  (void)in_sizes; (void)n_in; (void)out_size; (void)ws_size;
  const float* x      = (const float*)d_in[0];
  const float* vfirst = (const float*)d_in[1];
  const float* mr = (const float*)d_in[2];
  const float* mw = (const float*)d_in[3];
  const float* mk = (const float*)d_in[4];
  const float* mv = (const float*)d_in[5];
  const float* ma = (const float*)d_in[6];
  const float* mg = (const float*)d_in[7];
  const float* k_k = (const float*)d_in[8];
  const float* k_a = (const float*)d_in[9];
  const float* r_k = (const float*)d_in[10];
  const float* W_r = (const float*)d_in[11];
  const float* W_k = (const float*)d_in[12];
  const float* W_v = (const float*)d_in[13];
  const float* W_o = (const float*)d_in[14];
  const float* wA  = (const float*)d_in[15];
  const float* wB  = (const float*)d_in[16];
  const float* wbias = (const float*)d_in[17];
  const float* aA  = (const float*)d_in[18];
  const float* aB  = (const float*)d_in[19];
  const float* abias = (const float*)d_in[20];
  const float* vA  = (const float*)d_in[21];
  const float* vB  = (const float*)d_in[22];
  const float* vbias = (const float*)d_in[23];
  const float* gA  = (const float*)d_in[24];
  const float* gB  = (const float*)d_in[25];
  const float* gnw = (const float*)d_in[26];
  const float* gnb = (const float*)d_in[27];

  char* ws = (char*)d_ws;
  size_t off = 0;
  auto alloc = [&](size_t bytes) -> char* {
    char* p = ws + off;
    off += (bytes + 255) & ~(size_t)255;
    return p;
  };

  const size_t NBH = (size_t)BT_ * H_;
  // f16 activations / weights
  _Float16* xr16 = (_Float16*)alloc(NBH * 2);
  _Float16* xw16 = (_Float16*)alloc(NBH * 2);
  _Float16* xk16 = (_Float16*)alloc(NBH * 2);
  _Float16* xv16 = (_Float16*)alloc(NBH * 2);
  _Float16* xa16 = (_Float16*)alloc(NBH * 2);
  _Float16* xg16 = (_Float16*)alloc(NBH * 2);
  _Float16* Wr16 = (_Float16*)alloc((size_t)H_ * H_ * 2);
  _Float16* Wk16 = (_Float16*)alloc((size_t)H_ * H_ * 2);
  _Float16* Wv16 = (_Float16*)alloc((size_t)VD_ * H_ * 2);
  _Float16* Wo16 = (_Float16*)alloc((size_t)H_ * VD_ * 2);
  _Float16* wA16 = (_Float16*)alloc((size_t)64 * H_ * 2);
  _Float16* wB16 = (_Float16*)alloc((size_t)H_ * 64 * 2);
  _Float16* aA16 = (_Float16*)alloc((size_t)64 * H_ * 2);
  _Float16* aB16 = (_Float16*)alloc((size_t)H_ * 64 * 2);
  _Float16* vA16 = (_Float16*)alloc((size_t)64 * H_ * 2);
  _Float16* vB16 = (_Float16*)alloc((size_t)VD_ * 64 * 2);
  _Float16* gA16 = (_Float16*)alloc((size_t)160 * H_ * 2);
  _Float16* gB16 = (_Float16*)alloc((size_t)VD_ * 160 * 2);
  _Float16* wh16 = (_Float16*)alloc((size_t)BT_ * 64 * 2);
  _Float16* ah16 = (_Float16*)alloc((size_t)BT_ * 64 * 2);
  _Float16* vh16 = (_Float16*)alloc((size_t)BT_ * 64 * 2);
  _Float16* gh16 = (_Float16*)alloc((size_t)BT_ * 160 * 2);
  _Float16* og16 = (_Float16*)alloc(NBH * 2);
  // f32 intermediates
  float* r_   = (float*)alloc(NBH * 4);
  float* kbuf = (float*)alloc(NBH * 4);   // k, then kmod in place
  float* vbuf = (float*)alloc(NBH * 4);   // vraw, then blended v in place
  float* wl   = (float*)alloc(NBH * 4);   // lora out, then log-decay w in place
  float* al   = (float*)alloc(NBH * 4);   // lora out, then a in place
  float* vl   = (float*)alloc(NBH * 4);
  float* gl   = (float*)alloc(NBH * 4);   // lora out, then g in place
  float* adir = (float*)alloc(NBH * 4);
  float* bdir = (float*)alloc(NBH * 4);
  float* o_   = (float*)alloc(NBH * 4);
  float* wh1  = (float*)alloc((size_t)BT_ * 64 * 4);
  float* ah1  = (float*)alloc((size_t)BT_ * 64 * 4);
  float* vh1  = (float*)alloc((size_t)BT_ * 64 * 4);
  float* gh1  = (float*)alloc((size_t)BT_ * 160 * 4);

  auto cvt = [&](const float* s, _Float16* d, size_t n) {
    cvt_f16_kernel<<<dim3((unsigned)((n + 255) / 256)), dim3(256), 0, stream>>>(
        s, d, (int)n);
  };
  cvt(W_r, Wr16, (size_t)H_ * H_);
  cvt(W_k, Wk16, (size_t)H_ * H_);
  cvt(W_v, Wv16, (size_t)VD_ * H_);
  cvt(W_o, Wo16, (size_t)H_ * VD_);
  cvt(wA, wA16, (size_t)64 * H_);
  cvt(wB, wB16, (size_t)H_ * 64);
  cvt(aA, aA16, (size_t)64 * H_);
  cvt(aB, aB16, (size_t)H_ * 64);
  cvt(vA, vA16, (size_t)64 * H_);
  cvt(vB, vB16, (size_t)VD_ * 64);
  cvt(gA, gA16, (size_t)160 * H_);
  cvt(gB, gB16, (size_t)VD_ * 160);

  prep_mix_kernel<<<dim3(BT_ * H_ / 256), dim3(256), 0, stream>>>(
      x, mr, mw, mk, mv, ma, mg, xr16, xw16, xk16, xv16, xa16, xg16);

  // big projections
  launch_gemm<4, 1024, 1024>(xr16, Wr16, r_,   BT_, stream);
  launch_gemm<4, 1024, 1024>(xk16, Wk16, kbuf, BT_, stream);
  launch_gemm<4, 1024, 1024>(xv16, Wv16, vbuf, BT_, stream);
  // lora stage 1
  launch_gemm<4, 64, 1024>(xw16, wA16, wh1, BT_, stream);
  launch_gemm<4, 64, 1024>(xa16, aA16, ah1, BT_, stream);
  launch_gemm<4, 64, 1024>(xv16, vA16, vh1, BT_, stream);
  launch_gemm<2, 160, 1024>(xg16, gA16, gh1, BT_, stream);
  // activation + cvt for stage 2
  act_cvt_f16_kernel<<<dim3(BT_ * 64 / 256), dim3(256), 0, stream>>>(wh1, wh16, BT_ * 64, 1);
  act_cvt_f16_kernel<<<dim3(BT_ * 64 / 256), dim3(256), 0, stream>>>(ah1, ah16, BT_ * 64, 0);
  act_cvt_f16_kernel<<<dim3(BT_ * 64 / 256), dim3(256), 0, stream>>>(vh1, vh16, BT_ * 64, 0);
  act_cvt_f16_kernel<<<dim3(BT_ * 160 / 256), dim3(256), 0, stream>>>(gh1, gh16, BT_ * 160, 0);
  // lora stage 2
  launch_gemm<4, 1024, 64>(wh16, wB16, wl, BT_, stream);
  launch_gemm<4, 1024, 64>(ah16, aB16, al, BT_, stream);
  launch_gemm<4, 1024, 64>(vh16, vB16, vl, BT_, stream);
  launch_gemm<4, 1024, 160>(gh16, gB16, gl, BT_, stream);

  epilogue_gates_kernel<<<dim3(BT_ * H_ / 256), dim3(256), 0, stream>>>(
      wl, al, vl, gl, vbuf, vfirst, wbias, abias, vbias);

  kk_dirs_kernel<<<dim3(BT_ * NH_ / 8), dim3(256), 0, stream>>>(
      kbuf, al, adir, bdir, k_k, k_a);

  rwkv7_recurrence_kernel<<<dim3(B_ * NH_), dim3(256), 0, stream>>>(
      r_, wl, kbuf, vbuf, adir, bdir, o_);

  norm_bonus_gate_kernel<<<dim3(BT_ * NH_ / 8), dim3(256), 0, stream>>>(
      o_, r_, kbuf, vbuf, gl, r_k, gnw, gnb, og16);

  launch_gemm<4, 1024, 1024>(og16, Wo16, (float*)d_out, BT_, stream);
}
